// CausalSelfAttention_34437047779585
// MI455X (gfx1250) — matrix operand order repaired and checked by reading, not compiled, not run on hardware
//
#include <hip/hip_runtime.h>

// ---------------------------------------------------------------------------
// CDNA5 (gfx1250) causal self-attention, bf16 WMMA everywhere.
// D=1024, H=16, Dh=64, B=4, S=2048.  Compute-bound (~137 GFLOP vs ~100MB HBM).
// All matmuls: v_wmma_f32_16x16x32_bf16. Weights are pre-transposed during the
// fp32->bf16 conversion so both GEMM operands stage into LDS with async b128
// copies (global_load_async_to_lds_b128 + double buffering, s_wait_asynccnt).
// ---------------------------------------------------------------------------

#define D_MODEL 1024
#define SEQ     2048
#define BATCH   4
#define NHEAD   16
#define HDIM    64
#define QKV_LD  (3 * D_MODEL)
#define SCALE   0.125f   // 1/sqrt(64)

typedef __bf16 bf16_t;
typedef bf16_t v16bf __attribute__((ext_vector_type(16)));
typedef float  v8f   __attribute__((ext_vector_type(8)));

union FragA {
    unsigned int u[8];
    v16bf        bf;
};

// ---- gfx1250 async global->LDS path (guarded; falls back to sync copy) ----
#if defined(__AMDGCN__)
#  if __has_builtin(__builtin_amdgcn_global_load_async_to_lds_b128)
#    define HAVE_ASYNC 1
#  endif
#endif
#ifndef HAVE_ASYNC
#  define HAVE_ASYNC 0
#endif

#if HAVE_ASYNC
typedef int v4i_vec __attribute__((vector_size(16)));
typedef __attribute__((address_space(1))) v4i_vec* g_v4i_ptr;   // global (AS1)
typedef __attribute__((address_space(3))) v4i_vec* l_v4i_ptr;   // LDS (AS3)
#endif

__device__ __forceinline__ void g2l_b128(void* lds, const void* g) {
#if HAVE_ASYNC
    // global (AS1) src, LDS (AS3) dst; LDS offset = low 32 bits of flat addr.
    __builtin_amdgcn_global_load_async_to_lds_b128(
        (g_v4i_ptr)(unsigned long long)g,
        (l_v4i_ptr)(unsigned int)(unsigned long long)lds,
        0, 0);
#else
    *reinterpret_cast<uint4*>(lds) = *reinterpret_cast<const uint4*>(g);
#endif
}

#if HAVE_ASYNC
#  define ASYNC_WAIT_ALL()  asm volatile("s_wait_asynccnt 0x0" ::: "memory")
#  define ASYNC_WAIT_4()    asm volatile("s_wait_asynccnt 0x4" ::: "memory")
#else
#  define ASYNC_WAIT_ALL()
#  define ASYNC_WAIT_4()
#endif

__device__ __forceinline__ unsigned short f2bf(float x) {
    unsigned int u = __builtin_bit_cast(unsigned int, x);
    u += 0x7FFFu + ((u >> 16) & 1u);           // round-to-nearest-even
    return (unsigned short)(u >> 16);
}

__device__ __forceinline__ float rmax16(float v) {
    v = fmaxf(v, __shfl_xor(v, 1, 32));
    v = fmaxf(v, __shfl_xor(v, 2, 32));
    v = fmaxf(v, __shfl_xor(v, 4, 32));
    v = fmaxf(v, __shfl_xor(v, 8, 32));
    return v;
}
__device__ __forceinline__ float rsum16(float v) {
    v += __shfl_xor(v, 1, 32);
    v += __shfl_xor(v, 2, 32);
    v += __shfl_xor(v, 4, 32);
    v += __shfl_xor(v, 8, 32);
    return v;
}

// ---------------------------------------------------------------------------
// fp32 -> bf16 (grid-stride), no transpose (activations)
// ---------------------------------------------------------------------------
__global__ void conv_f32_bf16(const float* __restrict__ in,
                              unsigned short* __restrict__ out, int n) {
    int i = blockIdx.x * blockDim.x + threadIdx.x;
    int stride = gridDim.x * blockDim.x;
    for (; i < n; i += stride) out[i] = f2bf(in[i]);
}

// ---------------------------------------------------------------------------
// fp32 [K][N] -> bf16 [N][K]  (weights, tiled 64x64 LDS transpose)
// ---------------------------------------------------------------------------
__global__ __launch_bounds__(256) void convT_f32_bf16(
    const float* __restrict__ in, unsigned short* __restrict__ out,
    int K, int N) {
    __shared__ float tile[64][65];
    const int k0 = blockIdx.x * 64;
    const int n0 = blockIdx.y * 64;
#pragma unroll
    for (int j = 0; j < 16; ++j) {
        int id = threadIdx.x + j * 256;
        int r = id >> 6, c = id & 63;
        tile[r][c] = in[(size_t)(k0 + r) * N + n0 + c];
    }
    __syncthreads();
#pragma unroll
    for (int j = 0; j < 16; ++j) {
        int id = threadIdx.x + j * 256;
        int rn = id >> 6, c = id & 63;
        out[(size_t)(n0 + rn) * K + k0 + c] = f2bf(tile[c][rn]);
    }
}

// ---------------------------------------------------------------------------
// Tiled bf16 WMMA GEMM:  C[M,N] = A[M,K] * Bt[N,K]^T   (Bt is N-major)
// Block tile 128x128, K-step 32, 256 threads = 8 waves as 4(M) x 2(N);
// each wave owns 32x64 = 2x4 WMMA accumulators (8 wmma / K-step / wave).
// Double-buffered LDS, async global->LDS staging.
// ---------------------------------------------------------------------------
__global__ __launch_bounds__(256) void gemm_bf16_wmma(
    const unsigned short* __restrict__ A,
    const unsigned short* __restrict__ Bt,
    void* __restrict__ C,
    int N, int K, int out_is_f32) {

    __shared__ __align__(16) unsigned short As[2][128 * 32];  // [row][k]
    __shared__ __align__(16) unsigned short Bs[2][128 * 32];  // [col][k]

    const int t    = threadIdx.x;
    const int lane = t & 31;
    const int wave = t >> 5;
    const int wm   = wave >> 1;          // 0..3 (M)
    const int wn   = wave & 1;           // 0..1 (N)
    const int n    = lane & 15;
    const int half = lane >> 4;

    const int row_blk = blockIdx.y * 128;
    const int col_blk = blockIdx.x * 128;

    const unsigned short* Ag = A  + (size_t)row_blk * K;
    const unsigned short* Bg = Bt + (size_t)col_blk * K;

    // per-thread staging addresses: 2 b128 for A + 2 b128 for B per K-step
    const int sr = t >> 2;               // 0..63
    const int sc = (t & 3) * 8;          // 0,8,16,24

    v8f acc[2][4];
#pragma unroll
    for (int i = 0; i < 2; ++i)
#pragma unroll
        for (int j = 0; j < 4; ++j) acc[i][j] = {};

    const int nsteps = K >> 5;

    // prologue: stage K-tile 0 into buffer 0
#pragma unroll
    for (int i = 0; i < 2; ++i) {
        g2l_b128(&As[0][(sr + i * 64) * 32 + sc], Ag + (size_t)(sr + i * 64) * K + sc);
        g2l_b128(&Bs[0][(sr + i * 64) * 32 + sc], Bg + (size_t)(sr + i * 64) * K + sc);
    }

    for (int it = 0; it < nsteps; ++it) {
        const int p = it & 1;
        if (it + 1 < nsteps) {
            const int kb = (it + 1) * 32;
#pragma unroll
            for (int i = 0; i < 2; ++i) {
                g2l_b128(&As[p ^ 1][(sr + i * 64) * 32 + sc],
                         Ag + (size_t)(sr + i * 64) * K + kb + sc);
                g2l_b128(&Bs[p ^ 1][(sr + i * 64) * 32 + sc],
                         Bg + (size_t)(sr + i * 64) * K + kb + sc);
            }
            ASYNC_WAIT_4();   // wait for buffer p only (loads complete in order)
        } else {
            ASYNC_WAIT_ALL();
        }
        __syncthreads();

        // A fragments (contiguous k = 16*half + s, same as B): M = lane&15
        FragA af[2];
#pragma unroll
        for (int sm = 0; sm < 2; ++sm) {
            int row0 = wm * 32 + sm * 16 + n;
#pragma unroll
            for (int v = 0; v < 8; ++v) {
                int kk = half * 16 + v * 2;
                af[sm].u[v] =
                    *reinterpret_cast<const unsigned int*>(&As[p][row0 * 32 + kk]);
            }
        }
        // B fragments one at a time (keeps registers low), 2 wmma each
#pragma unroll
        for (int sn = 0; sn < 4; ++sn) {
            int col0 = wn * 64 + sn * 16 + n;
            FragA bfr;
#pragma unroll
            for (int v = 0; v < 8; ++v) {
                int kk = half * 16 + v * 2;
                bfr.u[v] =
                    *reinterpret_cast<const unsigned int*>(&Bs[p][col0 * 32 + kk]);
            }
            acc[0][sn] = __builtin_amdgcn_wmma_f32_16x16x32_bf16(
                false, af[0].bf, false, bfr.bf, (short)0, acc[0][sn], false, false);
            acc[1][sn] = __builtin_amdgcn_wmma_f32_16x16x32_bf16(
                false, af[1].bf, false, bfr.bf, (short)0, acc[1][sn], false, false);
        }
        __syncthreads();
    }

    // epilogue (C/D layout: lane holds col n, rows r+8*half)
#pragma unroll
    for (int sm = 0; sm < 2; ++sm)
#pragma unroll
        for (int sn = 0; sn < 4; ++sn)
#pragma unroll
            for (int r = 0; r < 8; ++r) {
                int row = row_blk + wm * 32 + sm * 16 + r + 8 * half;
                int col = col_blk + wn * 64 + sn * 16 + n;
                float val = acc[sm][sn][r];
                if (out_is_f32)
                    ((float*)C)[(size_t)row * N + col] = val;
                else
                    ((unsigned short*)C)[(size_t)row * N + col] = f2bf(val);
            }
}

// ---------------------------------------------------------------------------
// Flash attention: 4 waves per block = 4 consecutive 16-query tiles sharing
// one (b, h) and one cooperatively-staged V chunk. Uniform chunk loop bound
// (no divergent barriers; extra chunks for earlier tiles are fully masked and
// numerically inert). Online softmax per wave.
// qkv: [B, S, 3*D] bf16.  ctx: [B, S, D] bf16.
// ---------------------------------------------------------------------------
__global__ __launch_bounds__(128) void attn_wmma(
    const unsigned short* __restrict__ qkv,
    unsigned short* __restrict__ ctx) {

    __shared__ __align__(16) unsigned short Ps[4][16 * 32]; // per-wave P tile
    __shared__ __align__(16) unsigned short Vs[32 * 64];    // shared V chunk [key][d]

    const int t    = threadIdx.x;
    const int lane = t & 31;
    const int wave = t >> 5;            // 0..3
    const int n    = lane & 15;
    const int half = lane >> 4;
    const int qt   = blockIdx.x * 4 + wave;   // query tile (0..127)
    const int h    = blockIdx.y;
    const int b    = blockIdx.z;

    const size_t bbase = (size_t)b * SEQ * QKV_LD;
    const int    hoff  = h * HDIM;

    // ---- Q as two 16x32 A-fragments ----
    FragA qa[2];
    {
        const unsigned short* qrow =
            qkv + bbase + (size_t)(qt * 16 + n) * QKV_LD + hoff;
#pragma unroll
        for (int f = 0; f < 2; ++f)
#pragma unroll
            for (int v = 0; v < 8; ++v) {
                int kk = f * 32 + half * 16 + v * 2;
                qa[f].u[v] = *reinterpret_cast<const unsigned int*>(qrow + kk);
            }
    }

    v8f o[4];
#pragma unroll
    for (int dt = 0; dt < 4; ++dt) o[dt] = {};
    float mrow[8], lrow[8];
#pragma unroll
    for (int r = 0; r < 8; ++r) { mrow[r] = -3.0e38f; lrow[r] = 0.0f; }

    // uniform bound across the block: keys needed by the LAST tile in block
    const int kend = (blockIdx.x + 1) * 64;
    for (int k0 = 0; k0 < kend; k0 += 32) {
        // ---- cooperative V staging: 32 keys x 64 d, 2 async b128/thread ----
#pragma unroll
        for (int i = 0; i < 2; ++i) {
            int id  = t + i * 128;
            int key = id >> 3;            // 0..31
            int seg = (id & 7) * 8;       // 0..56
            g2l_b128(&Vs[key * 64 + seg],
                     qkv + bbase + (size_t)(k0 + key) * QKV_LD + 2 * D_MODEL + hoff + seg);
        }

        // ---- K^T B-fragments: contiguous 32B per lane per fragment ----
        FragA kbf[2][2];
#pragma unroll
        for (int tt = 0; tt < 2; ++tt) {
            const unsigned short* krow =
                qkv + bbase + (size_t)(k0 + tt * 16 + n) * QKV_LD + D_MODEL + hoff;
#pragma unroll
            for (int f = 0; f < 2; ++f) {
                const uint4* p = reinterpret_cast<const uint4*>(krow + f * 32 + half * 16);
                *reinterpret_cast<uint4*>(&kbf[tt][f].u[0]) = p[0];
                *reinterpret_cast<uint4*>(&kbf[tt][f].u[4]) = p[1];
            }
        }

        // ---- S = Q K^T (two 16x16 key tiles) ----
        float sv[2][8];
#pragma unroll
        for (int tt = 0; tt < 2; ++tt) {
            v8f s = {};
            s = __builtin_amdgcn_wmma_f32_16x16x32_bf16(
                false, qa[0].bf, false, kbf[tt][0].bf, (short)0, s, false, false);
            s = __builtin_amdgcn_wmma_f32_16x16x32_bf16(
                false, qa[1].bf, false, kbf[tt][1].bf, (short)0, s, false, false);
            int key = k0 + tt * 16 + n;
#pragma unroll
            for (int r = 0; r < 8; ++r) {
                int qg = qt * 16 + r + 8 * half;
                float x = s[r] * SCALE;
                sv[tt][r] = (key <= qg) ? x : -3.0e38f;
            }
        }

        // ---- online softmax (rows live at (r, half-group); 16-lane reduce) ----
#pragma unroll
        for (int r = 0; r < 8; ++r) {
            float mx    = rmax16(fmaxf(sv[0][r], sv[1][r]));
            float mnew  = fmaxf(mrow[r], mx);
            float alpha = __expf(mrow[r] - mnew);
            float p0    = __expf(sv[0][r] - mnew);
            float p1    = __expf(sv[1][r] - mnew);
            Ps[wave][(r + 8 * half) * 32 + n]      = f2bf(p0);
            Ps[wave][(r + 8 * half) * 32 + 16 + n] = f2bf(p1);
            lrow[r] = lrow[r] * alpha + rsum16(p0 + p1);
            mrow[r] = mnew;
#pragma unroll
            for (int dt = 0; dt < 4; ++dt) o[dt][r] *= alpha;
        }

        ASYNC_WAIT_ALL();
        __syncthreads();

        // ---- P A-fragment from own wave's LDS tile ----
        FragA pa;
#pragma unroll
        for (int v = 0; v < 8; ++v) {
            int kk = half * 16 + v * 2;
            pa.u[v] = *reinterpret_cast<const unsigned int*>(&Ps[wave][n * 32 + kk]);
        }
        // ---- O += P V  (4 d-tiles of 16) ----
#pragma unroll
        for (int dt = 0; dt < 4; ++dt) {
            FragA vb;
#pragma unroll
            for (int v = 0; v < 8; ++v) {
                int kl = half * 16 + v * 2;
                unsigned int lo = Vs[kl * 64 + dt * 16 + n];
                unsigned int hi = Vs[(kl + 1) * 64 + dt * 16 + n];
                vb.u[v] = lo | (hi << 16);
            }
            o[dt] = __builtin_amdgcn_wmma_f32_16x16x32_bf16(
                false, pa.bf, false, vb.bf, (short)0, o[dt], false, false);
        }
        __syncthreads();
    }

    // ---- normalize + store bf16 context ----
#pragma unroll
    for (int dt = 0; dt < 4; ++dt)
#pragma unroll
        for (int r = 0; r < 8; ++r) {
            int row = qt * 16 + r + 8 * half;
            int d   = dt * 16 + n;
            ctx[(size_t)b * SEQ * D_MODEL + (size_t)row * D_MODEL + hoff + d] =
                f2bf(o[dt][r] / lrow[r]);
        }
}

// ---------------------------------------------------------------------------
// Launcher
// ---------------------------------------------------------------------------
extern "C" void kernel_launch(void* const* d_in, const int* in_sizes, int n_in,
                              void* d_out, int out_size, void* d_ws, size_t ws_size,
                              hipStream_t stream) {
    (void)in_sizes; (void)n_in; (void)out_size; (void)ws_size;

    const float* x     = (const float*)d_in[0];  // [4,2048,1024]
    const float* w_qkv = (const float*)d_in[1];  // [1024,3072]
    const float* w_out = (const float*)d_in[2];  // [1024,1024]
    float*       out   = (float*)d_out;          // [4,2048,1024]

    // workspace layout (bytes): ~92 MB total
    char* ws = (char*)d_ws;
    unsigned short* xb    = (unsigned short*)(ws);                 // 16 MB  [8192,1024]
    unsigned short* wqkvT = (unsigned short*)(ws + 16777216);      //  6 MB  [3072,1024]
    unsigned short* woutT = (unsigned short*)(ws + 23068672);      //  2 MB  [1024,1024]
    unsigned short* qkvb  = (unsigned short*)(ws + 25165824);      // 48 MB  [8192,3072]
    unsigned short* ctxb  = (unsigned short*)(ws + 75497472);      // 16 MB  [8192,1024]

    const int NX = BATCH * SEQ * D_MODEL;          // 8388608
    const int M  = BATCH * SEQ;                    // 8192

    conv_f32_bf16<<<4096, 256, 0, stream>>>(x, xb, NX);
    // weights stored transposed: Wt[N][K]
    convT_f32_bf16<<<dim3(D_MODEL / 64, 3 * D_MODEL / 64), 256, 0, stream>>>(
        w_qkv, wqkvT, D_MODEL, 3 * D_MODEL);
    convT_f32_bf16<<<dim3(D_MODEL / 64, D_MODEL / 64), 256, 0, stream>>>(
        w_out, woutT, D_MODEL, D_MODEL);

    // qkv = x @ w_qkv   : [8192,1024] x [1024,3072] -> bf16
    gemm_bf16_wmma<<<dim3(3 * D_MODEL / 128, M / 128), 256, 0, stream>>>(
        xb, wqkvT, (void*)qkvb, 3 * D_MODEL, D_MODEL, 0);

    // flash attention: 4 query tiles (4 waves) per block
    attn_wmma<<<dim3(SEQ / 64, NHEAD, BATCH), 128, 0, stream>>>(qkvb, ctxb);

    // out = ctx @ w_out : [8192,1024] x [1024,1024] -> f32
    gemm_bf16_wmma<<<dim3(D_MODEL / 128, M / 128), 256, 0, stream>>>(
        ctxb, woutT, (void*)out, D_MODEL, D_MODEL, 1);
}
